// FGW_Layer_60662118089062
// MI455X (gfx1250) — compile-verified, matrix operand order checked by reading, not run on hardware
//
#include <hip/hip_runtime.h>
#include <hip/hip_bf16.h>

typedef __attribute__((ext_vector_type(16))) _Float16 v16h;
typedef __attribute__((ext_vector_type(8)))  float    v8f;

namespace {
constexpr int kNS = 16, kNF = 16, kD = 128, kK = 64;
constexpr int kNNodes = 100000, kAH = 32;
constexpr float kEps = 0.05f;
}

__device__ __forceinline__ float xorf(float v, int m) { return __shfl_xor(v, m, 32); }

// ---------------------------------------------------------------------------
// Prep: one block per k. Builds symmetric relu C_filter slice, c2[k][j],
// ff[j][k], and the f16 WMMA B-layout fragments for C (16x16, K-padded to 32)
// and F (128x16 split into 4 K=32 steps).
// B-frag layout (ISA 7.12.2): lane L holds col N=L%16; element e<8 ->
// K = (L/16)*8 + e, e>=8 -> K = 16 + (L/16)*8 + (e-8).
// ---------------------------------------------------------------------------
__global__ __launch_bounds__(256)
void fgw_prep(const float* __restrict__ C_up, const float* __restrict__ F,
              const float* __restrict__ hlog,
              float* __restrict__ c2w, float* __restrict__ ffw,
              _Float16* __restrict__ Cb, _Float16* __restrict__ Fb)
{
  const int k = blockIdx.x;
  const int t = threadIdx.x;
  __shared__ float Cf[kNF * kNF];
  __shared__ float hf[kNF];

  if (t == 0) {
    float mx = -1e30f;
    for (int i = 0; i < kNF; ++i) mx = fmaxf(mx, hlog[i]);
    float s = 0.f;
    for (int i = 0; i < kNF; ++i) { float e = __expf(hlog[i] - mx); hf[i] = e; s += e; }
    const float inv = 1.f / s;
    for (int i = 0; i < kNF; ++i) hf[i] *= inv;
  }
  {
    const int j = t >> 4, l = t & 15;
    float cv = 0.f;
    if (j != l) {
      const int a = j < l ? j : l;
      const int b2 = j < l ? l : j;
      const int p = a * (2 * kNF - a - 1) / 2 + (b2 - a - 1);   // triu index
      cv = fmaxf(C_up[p * kK + k], 0.f);
    }
    Cf[t] = cv;
  }
  __syncthreads();

  if (t < kNF) {                       // c2[k][j] = sum_l C[j][l]^2 * h_fil[l]
    float s = 0.f;
    for (int l = 0; l < kNF; ++l) { const float c = Cf[t * kNF + l]; s += c * c * hf[l]; }
    c2w[k * kNF + t] = s;
  } else if (t >= 32 && t < 64) {      // Cb B-frag: B[K=m][N=j] = C[j][m]
    const int lane = t - 32, col = lane & 15, half = lane >> 4;
    #pragma unroll
    for (int e = 0; e < 16; ++e)
      Cb[(k * 32 + lane) * 16 + e] =
          (e < 8) ? (_Float16)Cf[col * kNF + half * 8 + e] : (_Float16)0.f;
  } else if (t >= 64 && t < 80) {      // ff[j][k] = sum_d F[j][d][k]^2
    const int j = t - 64;
    float s = 0.f;
    for (int d = 0; d < kD; ++d) { const float f = F[(j * kD + d) * kK + k]; s += f * f; }
    ffw[j * kK + k] = s;
  }

  // Fb B-frags: 4 steps of K=32. B[K=d][N=j] = F[j][d][k]
  for (int f = t; f < 4 * 32 * 16; f += 256) {
    const int step = f >> 9, rem = f & 511, lane = rem >> 4, e = rem & 15;
    const int col = lane & 15, half = lane >> 4;
    const int d = step * 32 + (e < 8 ? half * 8 + e : 16 + half * 8 + (e - 8));
    Fb[((k * 4 + step) * 32 + lane) * 16 + e] = (_Float16)F[(col * kD + d) * kK + k];
  }
}

// ---------------------------------------------------------------------------
// Main: one block per batch element b, 8 waves; each wave owns 8 k-tiles.
// ---------------------------------------------------------------------------
__global__ __launch_bounds__(256)
void fgw_main(const float* __restrict__ adj, const float* __restrict__ features,
              const int* __restrict__ idxs,
              const float* __restrict__ aw1, const float* __restrict__ ab1,
              const float* __restrict__ aw2, const float* __restrict__ ab2,
              const float* __restrict__ hlog,
              const float* __restrict__ c2w, const float* __restrict__ ffw,
              const _Float16* __restrict__ Cb, const _Float16* __restrict__ Fb,
              float* __restrict__ out)
{
  __shared__ float xs[kNS * kD];
  __shared__ float adjs[kNS * kNS];
  __shared__ float xx[kNS], c1s[kNS], hf[kNS], xp[kD], hid[kAH];
  __shared__ int idxv[kNS];
  __shared__ float sAlpha;
  __shared__ _Float16 tr[8 * 256];     // per-wave 16x16 f16 transpose scratch

  const int tid = threadIdx.x;
  const int b = blockIdx.x;
  const int lane = tid & 31;
  const int wave = tid >> 5;
  const int col = lane & 15;           // N index of this lane in C/D frags
  const int half = lane >> 4;

  if (tid < kNS) idxv[tid] = idxs[b * kNS + tid];
  if (tid == 0) {                      // softmax(h_logits)
    float mx = -1e30f;
    for (int i = 0; i < kNF; ++i) mx = fmaxf(mx, hlog[i]);
    float s = 0.f;
    for (int i = 0; i < kNF; ++i) { float e = __expf(hlog[i] - mx); hf[i] = e; s += e; }
    const float inv = 1.f / s;
    for (int i = 0; i < kNF; ++i) hf[i] *= inv;
  }
  __syncthreads();

  // gather x (padding row -> zeros) and adj tile
  for (int i = tid; i < kNS * kD; i += 256) {
    const int row = i >> 7;
    const int idx = idxv[row];
    xs[i] = (idx < kNNodes) ? features[idx * kD + (i & 127)] : 0.f;
  }
  for (int i = tid; i < kNS * kNS; i += 256) adjs[i] = adj[b * kNS * kNS + i];
  __syncthreads();

  if (tid < kNS) {                     // xx[i], c1[i]
    float sx = 0.f, sa = 0.f;
    for (int d = 0; d < kD; ++d) { const float v = xs[tid * kD + d]; sx += v * v; }
    for (int l = 0; l < kNS; ++l) { const float a = adjs[tid * kNS + l]; sa += a * a; }
    xx[tid] = sx; c1s[tid] = sa * (1.f / 16.f);
  }
  if (tid < kD) {                      // mean-pooled features
    float s = 0.f;
    for (int i = 0; i < kNS; ++i) s += xs[i * kD + tid];
    xp[tid] = s * (1.f / 16.f);
  }
  __syncthreads();
  if (tid < kAH) {                     // alpha MLP hidden layer
    float s = ab1[tid];
    for (int d = 0; d < kD; ++d) s += xp[d] * aw1[d * kAH + tid];
    hid[tid] = fmaxf(s, 0.f);
  }
  __syncthreads();
  if (tid == 0) {
    float s = ab2[0];
    for (int h = 0; h < kAH; ++h) s += hid[h] * aw2[h];
    sAlpha = 1.f / (1.f + __expf(-s));
  }
  __syncthreads();

  // ---- per-lane k-invariant fragments ----
  v16h xA[4];                          // A-frags of x: rows i, K=d (4 steps of 32)
  #pragma unroll
  for (int s = 0; s < 4; ++s) {
    #pragma unroll
    for (int e = 0; e < 16; ++e) {
      const int d = s * 32 + (e < 8 ? half * 8 + e : 16 + half * 8 + (e - 8));
      xA[s][e] = (_Float16)xs[col * kD + d];
    }
  }
  v16h adjA;                           // A-frag of adj: rows i, K=l (padded to 32)
  #pragma unroll
  for (int e = 0; e < 16; ++e)
    adjA[e] = (e < 8) ? (_Float16)adjs[col * kNS + half * 8 + e] : (_Float16)0.f;

  float c1v[8], xxv[8];
  #pragma unroll
  for (int r = 0; r < 8; ++r) { c1v[r] = c1s[r + 8 * half]; xxv[r] = xx[r + 8 * half]; }
  const float hfl = hf[col];
  const float alpha = sAlpha;
  _Float16* tw = tr + wave * 256;
  const v8f z8 = {};

  // ---- k-invariant GW first step: A1 = adj @ T0, T0 = h_sub (x) h_fil ----
  // (independent of k: hoisted out of the loop, incl. its D->A transpose)
  v16h tb0;
  const _Float16 t0h = (_Float16)(hfl * 0.0625f);
  #pragma unroll
  for (int e = 0; e < 16; ++e) tb0[e] = (e < 8) ? t0h : (_Float16)0.f;
  v8f A1 = __builtin_amdgcn_wmma_f32_16x16x32_f16(false, adjA, false, tb0,
                                                  (short)0, z8, false, false);
  #pragma unroll
  for (int r = 0; r < 8; ++r) tw[(r + 8 * half) * 16 + col] = (_Float16)A1[r];
  __builtin_amdgcn_wave_barrier();
  v16h a1f;                            // A-frag of A1 (K = m, padded to 32)
  #pragma unroll
  for (int e = 0; e < 16; ++e)
    a1f[e] = (e < 8) ? tw[col * 16 + half * 8 + e] : (_Float16)0.f;
  __builtin_amdgcn_wave_barrier();

  for (int kk = 0; kk < 8; ++kk) {
    const int k = (kk << 3) + wave;
    if (kk < 7)
      __builtin_prefetch((const void*)(Fb + (size_t)(k + 8) * 4 * 32 * 16), 0, 1);

    const v16h cbf = *(const v16h*)(Cb + ((size_t)(k * 32 + lane) << 4));
    const float c2l = c2w[k * kNF + col];
    const float ffl = ffw[col * kK + k];

    // ---- cross[b,k] = x @ F_k (16x16, K=128): two independent WMMA chains ----
    v8f acc0 = {}, acc1 = {};
    {
      const v16h fb0 = *(const v16h*)(Fb + ((size_t)((k * 4 + 0) * 32 + lane) << 4));
      const v16h fb1 = *(const v16h*)(Fb + ((size_t)((k * 4 + 1) * 32 + lane) << 4));
      const v16h fb2 = *(const v16h*)(Fb + ((size_t)((k * 4 + 2) * 32 + lane) << 4));
      const v16h fb3 = *(const v16h*)(Fb + ((size_t)((k * 4 + 3) * 32 + lane) << 4));
      acc0 = __builtin_amdgcn_wmma_f32_16x16x32_f16(false, xA[0], false, fb0,
                                                    (short)0, acc0, false, false);
      acc1 = __builtin_amdgcn_wmma_f32_16x16x32_f16(false, xA[2], false, fb2,
                                                    (short)0, acc1, false, false);
      acc0 = __builtin_amdgcn_wmma_f32_16x16x32_f16(false, xA[1], false, fb1,
                                                    (short)0, acc0, false, false);
      acc1 = __builtin_amdgcn_wmma_f32_16x16x32_f16(false, xA[3], false, fb3,
                                                    (short)0, acc1, false, false);
    }
    // core1 = A1 @ Ck (A1 transposed once, outside the loop)
    v8f core = __builtin_amdgcn_wmma_f32_16x16x32_f16(false, a1f, false, cbf,
                                                      (short)0, z8, false, false);

    float Mreg[8], cC[8], cost[8];
    #pragma unroll
    for (int r = 0; r < 8; ++r) {
      Mreg[r] = xxv[r] + ffl - 2.f * (acc0[r] + acc1[r]);
      cC[r]   = c1v[r] + c2l;
      cost[r] = alpha * (cC[r] - 2.f * core[r]) + (1.f - alpha) * Mreg[r];
    }

    // tile-wide min (stabilized kernel), then exp
    float mn = cost[0];
    #pragma unroll
    for (int r = 1; r < 8; ++r) mn = fminf(mn, cost[r]);
    #pragma unroll
    for (int m = 1; m <= 16; m <<= 1) mn = fminf(mn, xorf(mn, m));
    float Km[8];
    #pragma unroll
    for (int r = 0; r < 8; ++r) Km[r] = __expf((mn - cost[r]) * (1.f / kEps));

    // ---- Sinkhorn (2 iters) entirely in registers ----
    float vj = 1.f, u[8];
    #pragma unroll
    for (int it = 0; it < 2; ++it) {
      #pragma unroll
      for (int r = 0; r < 8; ++r) {              // u = h_sub / (K v): row sums
        float t = Km[r] * vj;
        t += xorf(t, 1); t += xorf(t, 2); t += xorf(t, 4); t += xorf(t, 8);
        u[r] = 0.0625f / t;
      }
      float p = 0.f;                             // v = h_fil / (K^T u): col sums
      #pragma unroll
      for (int r = 0; r < 8; ++r) p += Km[r] * u[r];
      p += xorf(p, 16);
      vj = hfl / p;
    }
    float T[8];
    #pragma unroll
    for (int r = 0; r < 8; ++r) T[r] = u[r] * Km[r] * vj;

    // ---- final objective: recompute cost with final T ----
    v16h tb2;                                    // D-layout -> B-frag is per-lane
    #pragma unroll
    for (int e = 0; e < 16; ++e) tb2[e] = (e < 8) ? (_Float16)T[e] : (_Float16)0.f;

    v8f A2 = __builtin_amdgcn_wmma_f32_16x16x32_f16(false, adjA, false, tb2,
                                                    (short)0, z8, false, false);
    #pragma unroll
    for (int r = 0; r < 8; ++r) tw[(r + 8 * half) * 16 + col] = (_Float16)A2[r];
    __builtin_amdgcn_wave_barrier();
    v16h a2f;
    #pragma unroll
    for (int e = 0; e < 16; ++e)
      a2f[e] = (e < 8) ? tw[col * 16 + half * 8 + e] : (_Float16)0.f;
    __builtin_amdgcn_wave_barrier();

    v8f core2 = __builtin_amdgcn_wmma_f32_16x16x32_f16(false, a2f, false, cbf,
                                                       (short)0, z8, false, false);
    float obj = 0.f;
    #pragma unroll
    for (int r = 0; r < 8; ++r)
      obj += (alpha * (cC[r] - 2.f * core2[r]) + (1.f - alpha) * Mreg[r]) * T[r];
    obj += xorf(obj, 1); obj += xorf(obj, 2); obj += xorf(obj, 4);
    obj += xorf(obj, 8); obj += xorf(obj, 16);

    if (lane == 0) out[b * kK + k] = obj;
  }
}

// ---------------------------------------------------------------------------
extern "C" void kernel_launch(void* const* d_in, const int* in_sizes, int n_in,
                              void* d_out, int out_size, void* d_ws, size_t ws_size,
                              hipStream_t stream) {
  const float* adj      = (const float*)d_in[0];
  const float* features = (const float*)d_in[1];
  const int*   idxs     = (const int*)d_in[2];
  const float* C_up     = (const float*)d_in[3];
  const float* F_filter = (const float*)d_in[4];
  const float* h_logits = (const float*)d_in[5];
  const float* aw1      = (const float*)d_in[6];
  const float* ab1      = (const float*)d_in[7];
  const float* aw2      = (const float*)d_in[8];
  const float* ab2      = (const float*)d_in[9];

  char* ws = (char*)d_ws;
  float*    c2w = (float*)(ws);                         //  4 KB  [K][NF]
  float*    ffw = (float*)(ws + 4096);                  //  4 KB  [NF][K]
  _Float16* Cb  = (_Float16*)(ws + 8192);               // 64 KB  B-frags of C
  _Float16* Fb  = (_Float16*)(ws + 8192 + 65536);       // 256 KB B-frags of F

  fgw_prep<<<kK, 256, 0, stream>>>(C_up, F_filter, h_logits, c2w, ffw, Cb, Fb);
  fgw_main<<<2048, 256, 0, stream>>>(adj, features, idxs, aw1, ab1, aw2, ab2,
                                     h_logits, c2w, ffw, Cb, Fb, (float*)d_out);
}
